// GCN_51402168598673
// MI455X (gfx1250) — matrix operand order repaired and checked by reading, not compiled, not run on hardware
//
#include <hip/hip_runtime.h>
#include <hip/hip_bf16.h>

typedef __attribute__((ext_vector_type(16))) _Float16 v16h;
typedef __attribute__((ext_vector_type(8)))  float    v8f;

#define HID 256
#define IN_DIM 128
#define MB 64            // rows per GEMM block (4 panels of 16)

// ---------------------------------------------------------------------------
// Swizzled-activation index: panel-major, WMMA A-operand register order.
// For (row, k): lane = (rem>>3)&1 side, element i = (rem&7) + 8*(rem>>4).
// ---------------------------------------------------------------------------
__device__ __forceinline__ size_t asw_idx(int row, int k, int K) {
    int panel = row >> 4, r = row & 15;
    int c = k >> 5, rem = k & 31;
    int h2 = (rem >> 3) & 1;
    int i  = (rem & 7) + ((rem >> 4) << 3);
    return (size_t)panel * 16 * K + ((((c << 5) + (h2 << 4) + r) << 4) + i);
}

// ---------------------------------------------------------------------------
// Graph preprocessing
// ---------------------------------------------------------------------------
__global__ void k_cast_idx(const long long* __restrict__ ei, int* __restrict__ src,
                           int* __restrict__ dst, int E) {
    int e = blockIdx.x * blockDim.x + threadIdx.x;
    if (e < E) { src[e] = (int)ei[e]; dst[e] = (int)ei[(size_t)E + e]; }
}

__global__ void k_deg_init(float* __restrict__ deg, int N) {
    int n = blockIdx.x * blockDim.x + threadIdx.x;
    if (n < N) deg[n] = 1.0f;               // self-loop
}

__global__ void k_deg_acc(const int* __restrict__ dst, float* __restrict__ deg, int E) {
    int e = blockIdx.x * blockDim.x + threadIdx.x;
    if (e < E) atomicAdd(&deg[dst[e]], 1.0f);
}

__global__ void k_dinv(float* __restrict__ deg, int N) {
    int n = blockIdx.x * blockDim.x + threadIdx.x;
    if (n < N) deg[n] = rsqrtf(fmaxf(deg[n], 1.0f));
}

__global__ void k_norm(const int* __restrict__ src, const int* __restrict__ dst,
                       const float* __restrict__ dinv, float* __restrict__ norm, int E) {
    int e = blockIdx.x * blockDim.x + threadIdx.x;
    if (e < E) norm[e] = dinv[src[e]] * dinv[dst[e]];
}

// cast x (f32, row-major N x 128) -> Xp (f16, panel-swizzled)
__global__ void k_cast_x_sw(const float* __restrict__ x, _Float16* __restrict__ Xp, int N) {
    size_t t = (size_t)blockIdx.x * blockDim.x + threadIdx.x;
    int row = (int)(t >> 7);
    if (row >= N) return;
    int k = (int)(t & (IN_DIM - 1));
    Xp[asw_idx(row, k, IN_DIM)] = (_Float16)x[t];
}

// one-time weight swizzle: W (f32, K x OUT row-major) -> Wsw (f16, WMMA B order)
template <int K, int OUT>
__global__ void k_swizzle_W(const float* __restrict__ W, _Float16* __restrict__ Wsw) {
    constexpr int CH = K / 32;
    int s = blockIdx.x * blockDim.x + threadIdx.x;
    if (s >= K * OUT) return;
    int k   = s / OUT;
    int col = s - k * OUT;
    int ct  = col >> 4, c15 = col & 15;
    int c   = k >> 5,   rem = k & 31;
    int h2  = (rem >> 3) & 1, i   = (rem & 7) + ((rem >> 4) << 3);
    int d   = ((((ct * CH + c) << 5) + (h2 << 4) + c15) << 4) + i;
    Wsw[d] = (_Float16)W[s];
}

// ---------------------------------------------------------------------------
// WMMA GEMM: G[NP,OUT] = Xp[NP,K] @ Wsw[K,OUT] + bias
// block = 256 threads (8 waves); MB=64 rows (4 panels) x OUT=256 cols.
// A panel staged in static LDS (contiguous memcpy of pre-swizzled data);
// B read directly from L2-resident pre-swizzled global, held in registers
// and reused across the 4 row panels.
// ---------------------------------------------------------------------------
template <int K, int OUT>
__global__ void k_gemm_wmma(const _Float16* __restrict__ Xp, const _Float16* __restrict__ Wsw,
                            const float* __restrict__ bias, float* __restrict__ G) {
    static_assert(OUT == 256, "OUT must be 256");
    constexpr int CH = K / 32;
    constexpr int NP16 = MB / 16;                  // panels per block = 4
    __shared__ _Float16 At[MB * K];                // 16KB (K=128) / 32KB (K=256)

    const int tid  = threadIdx.x;
    const int lane = tid & 31;
    const int wv   = tid >> 5;
    const int l15  = lane & 15;
    const int hi   = lane >> 4;

    // contiguous A staging (pre-swizzled in global)
    {
        const uint4* s4 = (const uint4*)(Xp + (size_t)blockIdx.x * MB * K);
        uint4* d4 = (uint4*)At;
        for (int s = tid; s < MB * K / 8; s += blockDim.x) d4[s] = s4[s];
    }
    __syncthreads();

    const int ct0 = wv;
    const int ct1 = wv + 8;
    const v16h* Av = (const v16h*)At;              // K v16h per panel
    const v16h* Bv = (const v16h*)Wsw;

    v8f acc0[NP16] = {}, acc1[NP16] = {};

#pragma unroll
    for (int c = 0; c < CH; ++c) {
        v16h b0 = Bv[((ct0 * CH + c) << 5) + lane];
        v16h b1 = Bv[((ct1 * CH + c) << 5) + lane];
#pragma unroll
        for (int p = 0; p < NP16; ++p) {
            v16h a = Av[p * K + (c << 5) + lane];
            acc0[p] = __builtin_amdgcn_wmma_f32_16x16x32_f16(false, a, false, b0, (short)0, acc0[p], false, false);
            acc1[p] = __builtin_amdgcn_wmma_f32_16x16x32_f16(false, a, false, b1, (short)0, acc1[p], false, false);
        }
    }

    const int oc0 = ct0 * 16 + l15;
    const int oc1 = ct1 * 16 + l15;
    const float bias0 = bias[oc0];
    const float bias1 = bias[oc1];
#pragma unroll
    for (int p = 0; p < NP16; ++p) {
#pragma unroll
        for (int r = 0; r < 8; ++r) {
            int orow = blockIdx.x * MB + p * 16 + r + hi * 8;
            G[(size_t)orow * OUT + oc0] = acc0[p][r] + bias0;
            G[(size_t)orow * OUT + oc1] = acc1[p][r] + bias1;
        }
    }
}

// ---------------------------------------------------------------------------
// Aggregation: A[dst] += norm * G[src]; self-loop init A[i] = dinv[i]^2 * G[i]
// ---------------------------------------------------------------------------
__global__ void k_self_init(const float* __restrict__ dinv, const float* __restrict__ G,
                            float* __restrict__ A, int N) {
    size_t t = (size_t)blockIdx.x * blockDim.x + threadIdx.x;
    int n = (int)(t >> 6);
    if (n >= N) return;
    int f4 = (int)(t & 63) << 2;
    float w = dinv[n] * dinv[n];
    const float4 g = *(const float4*)(G + (size_t)n * HID + f4);
    float4 o; o.x = w * g.x; o.y = w * g.y; o.z = w * g.z; o.w = w * g.w;
    *(float4*)(A + (size_t)n * HID + f4) = o;
}

__global__ void k_edge_agg(const int* __restrict__ src, const int* __restrict__ dst,
                           const float* __restrict__ norm, const float* __restrict__ G,
                           float* __restrict__ A, int E) {
    size_t t = (size_t)blockIdx.x * blockDim.x + threadIdx.x;
    int e = (int)(t >> 6);
    if (e >= E) return;
    int f4 = (int)(t & 63) << 2;
    if ((t & 63) == 0) {
        // pull the sequential edge metadata ahead into cache (global_prefetch_b8)
        __builtin_prefetch(src + e + 4096, 0, 0);
        __builtin_prefetch(dst + e + 4096, 0, 0);
        __builtin_prefetch(norm + e + 4096, 0, 0);
    }
    int s = src[e], d = dst[e];
    float w = norm[e];
    const float4 g = *(const float4*)(G + (size_t)s * HID + f4);
    float* outp = A + (size_t)d * HID + f4;
    atomicAdd(outp + 0, w * g.x);
    atomicAdd(outp + 1, w * g.y);
    atomicAdd(outp + 2, w * g.z);
    atomicAdd(outp + 3, w * g.w);
}

// ---------------------------------------------------------------------------
// BatchNorm (training-mode, biased var) + ReLU -> swizzled f16 activations
// ---------------------------------------------------------------------------
__global__ void k_zero256(float* __restrict__ p) {
    int t = blockIdx.x * blockDim.x + threadIdx.x;
    if (t < 256) p[t] = 0.0f;
}

__global__ void k_bn_stats(const float* __restrict__ A, float* __restrict__ colsum,
                           float* __restrict__ colsq, int N) {
    int col = threadIdx.x;
    float s = 0.0f, sq = 0.0f;
    for (int r = blockIdx.x; r < N; r += gridDim.x) {
        float v = A[(size_t)r * HID + col];
        s += v; sq += v * v;
    }
    atomicAdd(&colsum[col], s);
    atomicAdd(&colsq[col], sq);
}

__global__ void k_bn_apply(const float* __restrict__ A, const float* __restrict__ colsum,
                           const float* __restrict__ colsq, const float* __restrict__ gamma,
                           const float* __restrict__ beta, _Float16* __restrict__ Xp, int N) {
    size_t t = (size_t)blockIdx.x * blockDim.x + threadIdx.x;
    int row = (int)(t >> 8);
    if (row >= N) return;
    int col = (int)(t & 255);
    float invN = 1.0f / (float)N;
    float mean = colsum[col] * invN;
    float var  = colsq[col] * invN - mean * mean;
    float v = (A[t] - mean) * rsqrtf(var + 1e-5f) * gamma[col] + beta[col];
    v = fmaxf(v, 0.0f);
    Xp[asw_idx(row, col, HID)] = (_Float16)v;
}

// ---------------------------------------------------------------------------
// Final projection (OUT=2) reading swizzled activations, + aggregation
// ---------------------------------------------------------------------------
__global__ void k_gemm_out2(const _Float16* __restrict__ Xp, const float* __restrict__ W2,
                            const float* __restrict__ b2, float* __restrict__ G2, int N) {
    int n = blockIdx.x * blockDim.x + threadIdx.x;
    if (n >= N) return;
    int panel = n >> 4, r = n & 15;
    const v16h* Xv = (const v16h*)Xp;              // HID v16h per panel
    float a0 = b2[0], a1 = b2[1];
#pragma unroll
    for (int c = 0; c < HID / 32; ++c) {
#pragma unroll
        for (int h2 = 0; h2 < 2; ++h2) {
            v16h v = Xv[(size_t)panel * HID + (c << 5) + (h2 << 4) + r];
#pragma unroll
            for (int i = 0; i < 16; ++i) {
                int k = c * 32 + (i & 7) + (h2 << 3) + ((i >> 3) << 4);
                float f = (float)v[i];
                a0 += f * W2[2 * k + 0];
                a1 += f * W2[2 * k + 1];
            }
        }
    }
    G2[2 * n + 0] = a0;
    G2[2 * n + 1] = a1;
}

__global__ void k_out_self(const float* __restrict__ dinv, const float* __restrict__ G2,
                           float* __restrict__ out, int N) {
    int n = blockIdx.x * blockDim.x + threadIdx.x;
    if (n >= N) return;
    float w = dinv[n] * dinv[n];
    out[2 * n + 0] = w * G2[2 * n + 0];
    out[2 * n + 1] = w * G2[2 * n + 1];
}

__global__ void k_out_edges(const int* __restrict__ src, const int* __restrict__ dst,
                            const float* __restrict__ norm, const float* __restrict__ G2,
                            float* __restrict__ out, int E) {
    int e = blockIdx.x * blockDim.x + threadIdx.x;
    if (e >= E) return;
    int s = src[e], d = dst[e];
    float w = norm[e];
    atomicAdd(&out[2 * d + 0], w * G2[2 * s + 0]);
    atomicAdd(&out[2 * d + 1], w * G2[2 * s + 1]);
}

// ---------------------------------------------------------------------------
// Host launcher
// ---------------------------------------------------------------------------
extern "C" void kernel_launch(void* const* d_in, const int* in_sizes, int n_in,
                              void* d_out, int out_size, void* d_ws, size_t ws_size,
                              hipStream_t stream) {
    const float*     x   = (const float*)d_in[0];
    const long long* ei  = (const long long*)d_in[1];
    const float*     W0  = (const float*)d_in[2];
    const float*     b0  = (const float*)d_in[3];
    const float*     W1  = (const float*)d_in[4];
    const float*     b1  = (const float*)d_in[5];
    const float*     W2  = (const float*)d_in[6];
    const float*     b2  = (const float*)d_in[7];
    const float*     g0  = (const float*)d_in[8];
    const float*     be0 = (const float*)d_in[9];
    const float*     g1  = (const float*)d_in[10];
    const float*     be1 = (const float*)d_in[11];
    float* out = (float*)d_out;

    const int N  = in_sizes[0] / IN_DIM;          // 100000
    const int E  = in_sizes[1] / 2;               // 1600000
    const int NB = (N + MB - 1) / MB;             // GEMM blocks
    const int NP = NB * MB;                       // padded rows

    char* base = (char*)d_ws;
    size_t off = 0;
    auto carve = [&](size_t bytes) -> char* {
        char* p = base + off;
        off = (off + bytes + 255) & ~(size_t)255;
        return p;
    };
    float*     dinv   = (float*)carve((size_t)N * 4);
    int*       src    = (int*)carve((size_t)E * 4);
    int*       dst    = (int*)carve((size_t)E * 4);
    float*     norm   = (float*)carve((size_t)E * 4);
    float*     G      = (float*)carve((size_t)NP * HID * 4);   // padded GEMM out
    float*     A      = (float*)carve((size_t)N * HID * 4);    // aggregated
    _Float16*  Xp     = (_Float16*)carve((size_t)NP * HID * 2); // padded swizzled acts
    _Float16*  Wsw0   = (_Float16*)carve((size_t)IN_DIM * HID * 2);
    _Float16*  Wsw1   = (_Float16*)carve((size_t)HID * HID * 2);
    float*     G2     = (float*)carve((size_t)N * 2 * 4);
    float*     colsum = (float*)carve(256 * 4);
    float*     colsq  = (float*)carve(256 * 4);
    (void)ws_size; (void)n_in; (void)out_size;

    const int T = 256;
    dim3 blk(T);
    auto cdiv = [](long long a, long long b) { return (unsigned)((a + b - 1) / b); };

    // graph norm + one-time weight swizzles
    k_cast_idx<<<cdiv(E, T), blk, 0, stream>>>(ei, src, dst, E);
    k_deg_init<<<cdiv(N, T), blk, 0, stream>>>(dinv, N);
    k_deg_acc<<<cdiv(E, T), blk, 0, stream>>>(dst, dinv, E);
    k_dinv<<<cdiv(N, T), blk, 0, stream>>>(dinv, N);
    k_norm<<<cdiv(E, T), blk, 0, stream>>>(src, dst, dinv, norm, E);
    k_swizzle_W<IN_DIM, HID><<<cdiv((long long)IN_DIM * HID, T), blk, 0, stream>>>(W0, Wsw0);
    k_swizzle_W<HID, HID><<<cdiv((long long)HID * HID, T), blk, 0, stream>>>(W1, Wsw1);

    // layer 0
    k_cast_x_sw<<<cdiv((long long)N * IN_DIM, T), blk, 0, stream>>>(x, Xp, N);
    k_gemm_wmma<IN_DIM, HID><<<NB, blk, 0, stream>>>(Xp, Wsw0, b0, G);
    k_self_init<<<cdiv((long long)N * 64, T), blk, 0, stream>>>(dinv, G, A, N);
    k_edge_agg<<<cdiv((long long)E * 64, T), blk, 0, stream>>>(src, dst, norm, G, A, E);
    k_zero256<<<1, blk, 0, stream>>>(colsum);
    k_zero256<<<1, blk, 0, stream>>>(colsq);
    k_bn_stats<<<512, blk, 0, stream>>>(A, colsum, colsq, N);
    k_bn_apply<<<cdiv((long long)N * HID, T), blk, 0, stream>>>(A, colsum, colsq, g0, be0, Xp, N);

    // layer 1
    k_gemm_wmma<HID, HID><<<NB, blk, 0, stream>>>(Xp, Wsw1, b1, G);
    k_self_init<<<cdiv((long long)N * 64, T), blk, 0, stream>>>(dinv, G, A, N);
    k_edge_agg<<<cdiv((long long)E * 64, T), blk, 0, stream>>>(src, dst, norm, G, A, E);
    k_zero256<<<1, blk, 0, stream>>>(colsum);
    k_zero256<<<1, blk, 0, stream>>>(colsq);
    k_bn_stats<<<512, blk, 0, stream>>>(A, colsum, colsq, N);
    k_bn_apply<<<cdiv((long long)N * HID, T), blk, 0, stream>>>(A, colsum, colsq, g1, be1, Xp, N);

    // layer 2
    k_gemm_out2<<<cdiv(N, T), blk, 0, stream>>>(Xp, W2, b2, G2, N);
    k_out_self<<<cdiv(N, T), blk, 0, stream>>>(dinv, G2, out, N);
    k_out_edges<<<cdiv(E, T), blk, 0, stream>>>(src, dst, norm, G2, out, E);
}
